// DifferentiableFeatureExtractor_1202590843033
// MI455X (gfx1250) — matrix operand order repaired and checked by reading, not compiled, not run on hardware
//
#include <hip/hip_runtime.h>

// ============================================================================
// MI455X (gfx1250) DiffEMA feature extractor, round 3.
//
//  * 20 truncated-geometric FIR EMAs as banded GEMMs on the WMMA pipe
//    (v_wmma_f32_16x16x32_f16, f16 hi/lo decomposition, f32 accumulate).
//  * All f32->f16 splits and Toeplitz weight tables precomputed ONCE, so the
//    FIR inner loop is pure loads + 3 WMMAs per 16-tap chunk.
//  * B operand staged workgroup-wide through LDS with
//    global_load_async_to_lds_b128 (ASYNCcnt) -- builtin confirmed present.
//  * Rolling windows: per-thread loops (12 MB inputs are L2-resident, 192 MB).
//  * cumsum/cummax scans: single-workgroup multi-pass scan.
//
// Workspace: 14 f32 slots + 32 f16 slots + A-tables  (~121 MB).
// ============================================================================

typedef __attribute__((ext_vector_type(16))) _Float16 v16h;
typedef __attribute__((ext_vector_type(8)))  _Float16 v8h;
typedef __attribute__((ext_vector_type(8)))  float    v8f;
typedef __attribute__((ext_vector_type(4)))  int      v4i;

__device__ __forceinline__ float nan0(float v) { return (v == v) ? v : 0.0f; }

#define HALO_MAX 576               // 16 * max nchunks (35) rounded up
#define TILE_WG  2048              // outputs per workgroup (8 waves x 256)

#if __has_builtin(__builtin_amdgcn_global_load_async_to_lds_b128)
#define HAS_ASYNC_LDS 1
typedef __attribute__((address_space(1))) void* as1_vp;
typedef __attribute__((address_space(3))) void* as3_vp;
typedef __attribute__((address_space(1))) v4i*  as1_v4i;
typedef __attribute__((address_space(3))) v4i*  as3_v4i;
#else
#define HAS_ASYNC_LDS 0
#endif

__device__ __forceinline__ void wait_async0() {
#if __has_builtin(__builtin_amdgcn_s_wait_asynccnt)
    __builtin_amdgcn_s_wait_asynccnt(0);
#else
    asm volatile("s_wait_asynccnt 0x0" ::: "memory");
#endif
}

// ---------------------------------------------------------------------------
// Split f32 -> (f16 hi, f16 lo)
// ---------------------------------------------------------------------------
__global__ __launch_bounds__(256)
void split_kernel(const float* __restrict__ x, _Float16* __restrict__ xh,
                  _Float16* __restrict__ xl, int T)
{
    int t = blockIdx.x * blockDim.x + threadIdx.x;
    if (t >= T) return;
    float v = x[t];
    _Float16 h = (_Float16)v;
    xh[t] = h;
    xl[t] = (_Float16)(v - (float)h);
}

// ---------------------------------------------------------------------------
// Build Toeplitz A-operand tables in this code's contiguous-k A order (16-bit 16x32):
//   A[m][k'] = w[16p + m - k' + 16],  valid iff 0 <= m-k'+16 < 16 and tap < K
// lane<16: row=lane,   k' {0..15}
// lane>=16: row=lane-16, k' {16..31}
// ---------------------------------------------------------------------------
__global__ __launch_bounds__(32)
void build_atab_kernel(_Float16* __restrict__ atabH, _Float16* __restrict__ atabL,
                       const float* __restrict__ w_alphas, int alpha_idx,
                       int K, int aoff)
{
    const int lane = threadIdx.x;          // 0..31
    const int p    = blockIdx.x;           // chunk
    const float alpha = 1.0f / (1.0f + expf(-w_alphas[alpha_idx]));
    const float l2om  = log2f(1.0f - alpha);
    const int am  = lane & 15;
    const int kA0 = (lane < 16) ? 0 : 16;
    const int kA1 = kA0 + 8;
    _Float16* dh = atabH + ((size_t)(aoff + p) * 32 + lane) * 16;
    _Float16* dl = atabL + ((size_t)(aoff + p) * 32 + lane) * 16;
    #pragma unroll
    for (int j = 0; j < 8; ++j) {
        int q1 = am - (kA0 + j) + 16;
        int q2 = am - (kA1 + j) + 16;
        int tap1 = 16 * p + q1;
        int tap2 = 16 * p + q2;
        float w1 = (q1 >= 0 && q1 < 16 && tap1 < K)
                     ? alpha * exp2f(l2om * (float)tap1) : 0.0f;
        float w2 = (q2 >= 0 && q2 < 16 && tap2 < K)
                     ? alpha * exp2f(l2om * (float)tap2) : 0.0f;
        _Float16 h1 = (_Float16)w1;
        _Float16 h2 = (_Float16)w2;
        dh[j]     = h1;  dl[j]     = (_Float16)(w1 - (float)h1);
        dh[8 + j] = h2;  dl[8 + j] = (_Float16)(w2 - (float)h2);
    }
}

// ---------------------------------------------------------------------------
// FIR EMA via WMMA.  y[t] = sum_{k<K} w[k] * x[max(t-k,0)]
// One wave => 256-output tile (t = t0 + 16n + m).  Per 16-tap chunk p:
// 16x32x16 GEMM with precomputed Toeplitz A and LDS-staged B windows.
// B[k'][n] = x[t0 + 16n - 16p - 16 + k']  (32-byte aligned accesses).
// ---------------------------------------------------------------------------
__global__ __launch_bounds__(256)
void fir_ema_wmma(const _Float16* __restrict__ xh, const _Float16* __restrict__ xl,
                  float* __restrict__ y, _Float16* __restrict__ yh,
                  _Float16* __restrict__ yl,
                  const _Float16* __restrict__ atabH,
                  const _Float16* __restrict__ atabL,
                  int aoff, int nchunks, int T)
{
    __shared__ __align__(32) _Float16 sxh[TILE_WG + HALO_MAX];
    __shared__ __align__(32) _Float16 sxl[TILE_WG + HALO_MAX];

    const int lane   = threadIdx.x & 31;
    const int wave   = threadIdx.x >> 5;
    const int wgBase = blockIdx.x * TILE_WG;
    const int halo   = 16 * nchunks;             // <= HALO_MAX
    const int total  = TILE_WG + halo;           // halves to stage (mult of 16)
    const int srcBase = wgBase - halo;

    // ---- stage x window (hi & lo) into LDS ----
    if (srcBase >= 0) {
#if HAS_ASYNC_LDS
        for (int i = threadIdx.x * 8; i < total; i += 256 * 8) {
            __builtin_amdgcn_global_load_async_to_lds_b128(
                (as1_v4i)(as1_vp)(xh + srcBase + i),
                (as3_v4i)(as3_vp)&sxh[i], 0, 0);
            __builtin_amdgcn_global_load_async_to_lds_b128(
                (as1_v4i)(as1_vp)(xl + srcBase + i),
                (as3_v4i)(as3_vp)&sxl[i], 0, 0);
        }
        wait_async0();
#else
        for (int i = threadIdx.x * 8; i < total; i += 256 * 8) {
            *(v8h*)&sxh[i] = *(const v8h*)&xh[srcBase + i];
            *(v8h*)&sxl[i] = *(const v8h*)&xl[srcBase + i];
        }
#endif
    } else {
        for (int i = threadIdx.x; i < total; i += 256) {
            int g = srcBase + i;
            if (g < 0) g = 0;                    // replicate-pad x[0]
            if (g > T - 1) g = T - 1;
            sxh[i] = xh[g];
            sxl[i] = xl[g];
        }
    }
    __syncthreads();

    const int t0 = wgBase + wave * 256;
    if (t0 >= T) return;

    // C/D layout: lane -> column n; vgpr r -> row m0+r
    const int n   = lane & 15;
    const int m0  = (lane < 16) ? 0 : 8;
    // B layout: lane holds column n; K half selected by lane half
    const int kB0 = (lane < 16) ? 0 : 16;
    // LDS offset of this lane's B window for chunk p: ldsOff - 16p
    const int ldsOff = wave * 256 + 16 * n + kB0 - 16 + halo;

    v8f acc = {0.f, 0.f, 0.f, 0.f, 0.f, 0.f, 0.f, 0.f};

    const _Float16* aH = atabH + ((size_t)aoff * 32 + lane) * 16;
    const _Float16* aL = atabL + ((size_t)aoff * 32 + lane) * 16;
    for (int p = 0; p < nchunks; ++p) {
        const v16h a_hi = *(const v16h*)(aH + (size_t)p * 32 * 16);
        const v16h a_lo = *(const v16h*)(aL + (size_t)p * 32 * 16);
        const int  li   = ldsOff - 16 * p;
        const v16h b_hi = *(const v16h*)&sxh[li];
        const v16h b_lo = *(const v16h*)&sxl[li];
        acc = __builtin_amdgcn_wmma_f32_16x16x32_f16(
                  false, a_hi, false, b_hi, (short)0, acc, false, false);
        acc = __builtin_amdgcn_wmma_f32_16x16x32_f16(
                  false, a_hi, false, b_lo, (short)0, acc, false, false);
        acc = __builtin_amdgcn_wmma_f32_16x16x32_f16(
                  false, a_lo, false, b_hi, (short)0, acc, false, false);
    }

    // ---- store: acc[r] -> t = t0 + 16n + m0 + r (8 contiguous) ----
    const int tout = t0 + 16 * n + m0;
    float4 lo4 = make_float4(acc[0], acc[1], acc[2], acc[3]);
    float4 hi4 = make_float4(acc[4], acc[5], acc[6], acc[7]);
    *(float4*)(y + tout)     = lo4;
    *(float4*)(y + tout + 4) = hi4;
    // emit f16 hi/lo pair for chained FIR consumers
    v8h oh, ol;
    #pragma unroll
    for (int r = 0; r < 8; ++r) {
        float v = acc[r];
        _Float16 h = (_Float16)v;
        oh[r] = h;
        ol[r] = (_Float16)(v - (float)h);
    }
    *(v8h*)(yh + tout) = oh;
    *(v8h*)(yl + tout) = ol;
}

// ---------------------------------------------------------------------------
// Rolling windows: hh/ll over {204,36,18,9} + population std over 18.
// rsv outputs emitted directly as f16 hi/lo (only FIR consumes them).
// ---------------------------------------------------------------------------
__global__ __launch_bounds__(256)
void rolling_kernel(const float* __restrict__ C, const float* __restrict__ H,
                    const float* __restrict__ L, float* __restrict__ dis,
                    _Float16* __restrict__ r204h, _Float16* __restrict__ r204l,
                    _Float16* __restrict__ r18h,  _Float16* __restrict__ r18l,
                    _Float16* __restrict__ r9h,   _Float16* __restrict__ r9l,
                    _Float16* __restrict__ r36h,  _Float16* __restrict__ r36l,
                    int T)
{
    int t = blockIdx.x * blockDim.x + threadIdx.x;
    if (t >= T) return;
    float hh204 = -3.4e38f, ll204 = 3.4e38f;
    float hh36  = -3.4e38f, ll36  = 3.4e38f;
    float hh18  = -3.4e38f, ll18  = 3.4e38f;
    float hh9   = -3.4e38f, ll9   = 3.4e38f;
    float s1 = 0.f, s2 = 0.f;
    const int jmax = (t < 203) ? t : 203;
    for (int j = 0; j <= jmax; ++j) {
        float hv = H[t - j], lv = L[t - j];
        hh204 = fmaxf(hh204, hv);  ll204 = fminf(ll204, lv);
        if (j < 36) { hh36 = fmaxf(hh36, hv);  ll36 = fminf(ll36, lv); }
        if (j < 18) {
            hh18 = fmaxf(hh18, hv);  ll18 = fminf(ll18, lv);
            float cv = C[t - j];  s1 += cv;  s2 += cv * cv;
        }
        if (j < 9)  { hh9  = fmaxf(hh9, hv);   ll9  = fminf(ll9, lv); }
    }
    float cnt18 = (float)(((t < 17) ? t : 17) + 1);
    float mu = s1 / cnt18;
    dis[t] = sqrtf(fmaxf(s2 / cnt18 - mu * mu, 0.0f));

    float c = C[t];
    #define EMIT(hh, ll, oh, ol)                                               \
    {   float r = fminf(fmaxf((c - (ll)) / fmaxf((hh) - (ll), 1e-8f) * 100.0f, \
                              0.0f), 100.0f);                                  \
        _Float16 h = (_Float16)r;                                              \
        oh[t] = h;  ol[t] = (_Float16)(r - (float)h); }
    EMIT(hh204, ll204, r204h, r204l)
    EMIT(hh36,  ll36,  r36h,  r36l)
    EMIT(hh18,  ll18,  r18h,  r18l)
    EMIT(hh9,   ll9,   r9h,   r9l)
    #undef EMIT
}

// ---------------------------------------------------------------------------
// TEMA combine: TEMA3/TEU3/TED + T1s/T2s/T3s (shifted ratios, nan_to_num).
// ---------------------------------------------------------------------------
__global__ __launch_bounds__(256)
void tema_kernel(const float* __restrict__ E1, const float* __restrict__ E2,
                 const float* __restrict__ E3, const float* __restrict__ P1,
                 const float* __restrict__ P2, const float* __restrict__ P3,
                 const float* __restrict__ dis,
                 float* __restrict__ TEU3, float* __restrict__ TEMA3,
                 float* __restrict__ TED,  float* __restrict__ T1s,
                 float* __restrict__ T2s,  float* __restrict__ T3s, int T)
{
    int t = blockIdx.x * blockDim.x + threadIdx.x;
    if (t >= T) return;
    int i1 = (t >= 1) ? t - 1 : 0;
    int i6 = (t >= 6) ? t - 6 : 0;
    float v  = 3.f * (E1[t]  - E2[t])  + E3[t];
    float r1 = 3.f * (E1[i1] - E2[i1]) + E3[i1];
    float r6 = 3.f * (E1[i6] - E2[i6]) + E3[i6];
    float d  = dis[t];
    TEMA3[t] = v;  TEU3[t] = v + d;  TED[t] = v - d;
    T3s[t] = nan0((v - r6) / (fabsf(r6) + 1e-8f));
    T1s[t] = nan0((v - r1) / (fabsf(r1) + 1e-8f));
    float p  = 3.f * P1[t]  - 3.f * P2[t]  + P3[t];
    float rp = 3.f * P1[i6] - 3.f * P2[i6] + P3[i6];
    T2s[t] = nan0((p - rp) / (fabsf(rp) + 1e-8f));
}

// ---------------------------------------------------------------------------
// J combines + JX_base/F1/F2 (f32 rows + f16 hi/lo for FIR consumers)
// ---------------------------------------------------------------------------
__global__ __launch_bounds__(256)
void jx_kernel(const float* __restrict__ K1, const float* __restrict__ D1,
               const float* __restrict__ K2, const float* __restrict__ D2,
               const float* __restrict__ K3, const float* __restrict__ D3,
               const float* __restrict__ KN, const float* __restrict__ DN,
               const float* __restrict__ T1s, const float* __restrict__ T2s,
               const float* __restrict__ T3s,
               float* __restrict__ J1, float* __restrict__ J2,
               float* __restrict__ J3, float* __restrict__ JN36,
               float* __restrict__ JXb, float* __restrict__ F1,
               float* __restrict__ F2,
               _Float16* __restrict__ jxh, _Float16* __restrict__ jxl,
               _Float16* __restrict__ f1h, _Float16* __restrict__ f1l,
               _Float16* __restrict__ f2h, _Float16* __restrict__ f2l, int T)
{
    int t = blockIdx.x * blockDim.x + threadIdx.x;
    if (t >= T) return;
    float j1 = 3.f * K1[t] - 2.f * D1[t];
    float j2 = 3.f * K2[t] - 2.f * D2[t];
    float j3 = 3.f * K3[t] - 2.f * D3[t];
    J1[t] = j1;  J2[t] = j2;  J3[t] = j3;
    JN36[t] = 3.f * KN[t] - 2.f * DN[t];
    float jb = j1 + j2 + j3 * T1s[t];
    float f1 = j2 * T3s[t];
    float f2 = j1 * T2s[t];
    JXb[t] = jb;  F1[t] = f1;  F2[t] = f2;
    _Float16 h;
    h = (_Float16)jb;  jxh[t] = h;  jxl[t] = (_Float16)(jb - (float)h);
    h = (_Float16)f1;  f1h[t] = h;  f1l[t] = (_Float16)(f1 - (float)h);
    h = (_Float16)f2;  f2h[t] = h;  f2l[t] = (_Float16)(f2 - (float)h);
}

__global__ __launch_bounds__(256)
void finjx_kernel(const float* __restrict__ JXb, const float* __restrict__ F1,
                  const float* __restrict__ F2,  const float* __restrict__ EJX,
                  const float* __restrict__ EF1, const float* __restrict__ EF2,
                  const float* __restrict__ E8JX,const float* __restrict__ E8F1,
                  const float* __restrict__ E8F2,
                  float* __restrict__ JX, float* __restrict__ EMAJX,
                  float* __restrict__ EMAJX8, int T)
{
    int t = blockIdx.x * blockDim.x + threadIdx.x;
    if (t >= T) return;
    JX[t]     = JXb[t]  - 50.f + F1[t]  * 6.f + F2[t]  * 6.f;
    EMAJX[t]  = EJX[t]  - 50.f + EF1[t] * 6.f + EF2[t] * 6.f;
    EMAJX8[t] = E8JX[t] - 50.f + E8F1[t]* 6.f + E8F2[t]* 6.f;
}

// ---------------------------------------------------------------------------
// Three sequential scans in one single-workgroup pass.
// ---------------------------------------------------------------------------
#define SCAN_T 1024
__global__ __launch_bounds__(SCAN_T)
void scan3_kernel(const float* __restrict__ C, const float* __restrict__ JX,
                  const float* __restrict__ EJ,
                  float* __restrict__ cs, int* __restrict__ lastUp,
                  int* __restrict__ lastDn, int T)
{
    __shared__ float sf[SCAN_T];
    __shared__ int   su[SCAN_T];
    __shared__ int   sd[SCAN_T];
    const int tid = threadIdx.x;
    float carry = 0.f;  int cu = -1, cd = -1;
    if (tid == 0) cs[0] = 0.f;
    for (int base = 0; base < T; base += SCAN_T) {
        int t  = base + tid;
        int tc = (t < T) ? t : T - 1;
        int tp = (tc > 0) ? tc - 1 : 0;
        float jx  = JX[tc], ej  = EJ[tc];
        float jxp = JX[tp], ejp = EJ[tp];
        bool up = (jx > ej) && (jxp <= ejp) && (t > 0);
        bool dn = (jx < ej) && (jxp >= ejp) && (t > 0);
        sf[tid] = (t < T) ? C[t] : 0.f;
        su[tid] = up ? t : -1;
        sd[tid] = dn ? t : -1;
        __syncthreads();
        for (int off = 1; off < SCAN_T; off <<= 1) {
            float pv = 0.f;  int pu = -1, pd = -1;
            if (tid >= off) { pv = sf[tid - off]; pu = su[tid - off]; pd = sd[tid - off]; }
            __syncthreads();
            sf[tid] += pv;
            su[tid] = (su[tid] > pu) ? su[tid] : pu;
            sd[tid] = (sd[tid] > pd) ? sd[tid] : pd;
            __syncthreads();
        }
        if (t < T) {
            cs[t + 1]  = carry + sf[tid];
            int u = su[tid];  lastUp[t] = (cu > u) ? cu : u;
            int d = sd[tid];  lastDn[t] = (cd > d) ? cd : d;
        }
        carry += sf[SCAN_T - 1];
        int u = su[SCAN_T - 1];  cu = (cu > u) ? cu : u;
        int d = sd[SCAN_T - 1];  cd = (cd > d) ? cd : d;
        __syncthreads();
    }
}

// ---------------------------------------------------------------------------
// Dynamic MA from cumsum + barslast.
// ---------------------------------------------------------------------------
__global__ __launch_bounds__(256)
void ma_dyn_kernel(const float* __restrict__ cs, const int* __restrict__ lastDn,
                   const int* __restrict__ lastUp,
                   float* __restrict__ rowDown, float* __restrict__ rowUp, int T)
{
    int t = blockIdx.x * blockDim.x + threadIdx.x;
    if (t >= T) return;
    #pragma unroll
    for (int which = 0; which < 2; ++which) {
        int last = which ? lastUp[t] : lastDn[t];
        int n = (last >= 0) ? (t - last) : 0;
        float v = 0.f;
        if (n > 0) {
            int start = t + 1 - n;
            if (start < 0) start = 0;
            if (start > T) start = T;
            int cnt = (t + 1 - start);  if (cnt < 1) cnt = 1;
            v = nan0((cs[t + 1] - cs[start]) / (float)cnt);
        }
        if (which) rowUp[t] = v; else rowDown[t] = v;
    }
}

// ===========================================================================
extern "C" void kernel_launch(void* const* d_in, const int* in_sizes, int n_in,
                              void* d_out, int out_size, void* d_ws, size_t ws_size,
                              hipStream_t stream)
{
    const float* C  = (const float*)d_in[0];
    const float* H  = (const float*)d_in[1];
    const float* L  = (const float*)d_in[2];
    const float* wA = (const float*)d_in[3];
    float* out = (float*)d_out;
    const int T = in_sizes[0];

    // ---- workspace layout ----
    const size_t SL  = (size_t)T + 64;                  // f32 slot stride
    const size_t SLH = (size_t)T + 64;                  // f16 slot stride
    float* W = (float*)d_ws;
    auto slot = [&](int i) { return W + (size_t)i * SL; };
    _Float16* Hb = (_Float16*)(W + 14 * SL);
    auto hpH = [&](int i) { return Hb + (size_t)(2 * i)     * SLH; };
    auto hpL = [&](int i) { return Hb + (size_t)(2 * i + 1) * SLH; };
    _Float16* atabH = Hb + 32 * SLH;
    _Float16* atabL = atabH + 160 * 32 * 16;            // >=148 chunks
    auto row  = [&](int r) { return out + (size_t)r * (size_t)T; };

    const dim3 bE(256);
    const dim3 gE((T + 255) / 256);
    const int  gF = (T + TILE_WG - 1) / TILE_WG;

    // ---- static FIR config (KS from reference _max_kernel) ----
    static const int KS_[20] = {72,72,72,286,286,286,559,89,54,47,40,30,
                                130,30,30,30,30,37,37,37};
    int nch[20], aoff[20], acc = 0;
    for (int i = 0; i < 20; ++i) {
        nch[i]  = (KS_[i] + 15) / 16;
        aoff[i] = acc;
        acc += nch[i];
    }

    // ---- build weight tables ----
    for (int i = 0; i < 20; ++i)
        build_atab_kernel<<<nch[i], 32, 0, stream>>>(atabH, atabL, wA, i,
                                                     KS_[i], aoff[i]);

    auto FIR = [&](const _Float16* xih, const _Float16* xil, float* yo,
                   _Float16* yoh, _Float16* yol, int sidx) {
        fir_ema_wmma<<<gF, 256, 0, stream>>>(xih, xil, yo, yoh, yol,
                                             atabH, atabL, aoff[sidx],
                                             nch[sidx], T);
    };

    // rows 0..2 : raw C, H, L
    (void)hipMemcpyAsync(row(0), C, (size_t)T * sizeof(float), hipMemcpyDeviceToDevice, stream);
    (void)hipMemcpyAsync(row(1), H, (size_t)T * sizeof(float), hipMemcpyDeviceToDevice, stream);
    (void)hipMemcpyAsync(row(2), L, (size_t)T * sizeof(float), hipMemcpyDeviceToDevice, stream);

    // split C once
    split_kernel<<<gE, bE, 0, stream>>>(C, hpH(0), hpL(0), T);

    // f16 pair ids: 0=C 1=EMA1 2=EMA2 3=E21 4=E221 5=r204 6=r18 7=r9 8=r36
    //               9=K1 10=kdj-tmp 11=JXb 12=F1 13=F2 14=trash
    FIR(hpH(0), hpL(0), slot(0), hpH(1),  hpL(1),  0);   // EMA1
    FIR(hpH(1), hpL(1), slot(1), hpH(2),  hpL(2),  1);   // EMA2
    FIR(hpH(2), hpL(2), slot(2), hpH(14), hpL(14), 2);   // EMA3
    FIR(hpH(0), hpL(0), slot(3), hpH(3),  hpL(3),  3);   // E21
    FIR(hpH(3), hpL(3), slot(4), hpH(4),  hpL(4),  4);   // E221
    FIR(hpH(4), hpL(4), slot(5), hpH(14), hpL(14), 5);   // E231

    rolling_kernel<<<gE, bE, 0, stream>>>(C, H, L, slot(6),
                                          hpH(5), hpL(5), hpH(6), hpL(6),
                                          hpH(7), hpL(7), hpH(8), hpL(8), T);

    tema_kernel<<<gE, bE, 0, stream>>>(slot(0), slot(1), slot(2),
                                       slot(3), slot(4), slot(5), slot(6),
                                       row(3), row(4), row(5),
                                       row(6), row(7), row(8), T);

    // KDJ chains
    FIR(hpH(5),  hpL(5),  row(9),  hpH(9),  hpL(9),  6);   // K1
    FIR(hpH(9),  hpL(9),  row(10), hpH(14), hpL(14), 7);   // D1
    FIR(hpH(6),  hpL(6),  row(12), hpH(10), hpL(10), 8);   // K2
    FIR(hpH(10), hpL(10), row(13), hpH(14), hpL(14), 9);   // D2
    FIR(hpH(7),  hpL(7),  row(15), hpH(10), hpL(10), 10);  // K3
    FIR(hpH(10), hpL(10), row(16), hpH(14), hpL(14), 11);  // D3
    FIR(hpH(8),  hpL(8),  slot(0), hpH(10), hpL(10), 12);  // KN3
    FIR(hpH(10), hpL(10), slot(1), hpH(14), hpL(14), 13);  // DN3

    jx_kernel<<<gE, bE, 0, stream>>>(row(9), row(10), row(12), row(13),
                                     row(15), row(16), slot(0), slot(1),
                                     row(6), row(7), row(8),
                                     row(11), row(14), row(17), row(18),
                                     row(19), row(20), row(21),
                                     hpH(11), hpL(11), hpH(12), hpL(12),
                                     hpH(13), hpL(13), T);

    FIR(hpH(11), hpL(11), row(22), hpH(14), hpL(14), 14);  // EMA_JX
    FIR(hpH(12), hpL(12), row(23), hpH(14), hpL(14), 15);  // EMA_F1
    FIR(hpH(13), hpL(13), row(24), hpH(14), hpL(14), 16);  // EMA_F2
    FIR(hpH(11), hpL(11), slot(2), hpH(14), hpL(14), 17);  // EMA8_JX
    FIR(hpH(12), hpL(12), slot(3), hpH(14), hpL(14), 18);  // EMA8_F1
    FIR(hpH(13), hpL(13), slot(4), hpH(14), hpL(14), 19);  // EMA8_F2

    finjx_kernel<<<gE, bE, 0, stream>>>(row(19), row(20), row(21),
                                        row(22), row(23), row(24),
                                        slot(2), slot(3), slot(4),
                                        row(27), row(28), row(29), T);

    scan3_kernel<<<1, SCAN_T, 0, stream>>>(C, row(27), row(28),
                                           slot(11), (int*)slot(12),
                                           (int*)slot(13), T);

    ma_dyn_kernel<<<gE, bE, 0, stream>>>(slot(11), (int*)slot(13),
                                         (int*)slot(12), row(25), row(26), T);
}
